// GAT_3195455668264
// MI455X (gfx1250) — matrix operand order repaired and checked by reading, not compiled, not run on hardware
//
#include <hip/hip_runtime.h>

// ---------------------------------------------------------------------------
// GAT 2-layer forward for MI455X (gfx1250), wave32 + WMMA bf16.
//   GEMMs:      v_wmma_f32_16x16x32_bf16 (bf16 in, f32 accumulate),
//               W pre-transposed so A and B fragments are contiguous
//               16B-aligned chunks -> global_load_b128 only.
//   Softmax/agg: CSR grouped by destination -> no float atomics, fused
//                head-mean + bias + relu, LDS-staged coefficients,
//                global_prefetch for the gather stream.
// ---------------------------------------------------------------------------

typedef __bf16 bf16;
typedef bf16  v8bf  __attribute__((ext_vector_type(8)));
typedef bf16  v16bf __attribute__((ext_vector_type(16)));
typedef float v8f   __attribute__((ext_vector_type(8)));

#define NEG_SLOPE 0.2f

// ---------------- fp32 -> bf16 conversion ----------------------------------
__global__ void cvt_f32_to_bf16(const float* __restrict__ in,
                                bf16* __restrict__ out, int n) {
    int i = blockIdx.x * blockDim.x + threadIdx.x;
    if (i < n) out[i] = (bf16)in[i];
}

// ---------------- fp32 W[K][N] -> bf16 Wt[N][K] (convert + transpose) -------
__global__ void cvt_transpose_w(const float* __restrict__ W,
                                bf16* __restrict__ Wt, int K, int N) {
    int idx = blockIdx.x * blockDim.x + threadIdx.x;
    if (idx >= K * N) return;
    int k = idx / N, n = idx - k * N;          // coalesced read along N
    Wt[(size_t)n * K + k] = (bf16)W[idx];
}

// ---------------- WMMA GEMM: C[MxN] = A[MxK] * B[KxN] -----------------------
// A row-major [M][K] bf16, B given transposed as Bt[N][K] bf16, C fp32.
// One wave computes a 16x32 strip (two 16x16 tiles share the A fragment).
// All fragment reads are contiguous 16B-aligned chunks -> global_load_b128.
__global__ void wmma_gemm_bf16(const bf16* __restrict__ A,
                               const bf16* __restrict__ Bt,
                               float* __restrict__ C,
                               int M, int K, int N) {
    const int wave = (blockIdx.x * blockDim.x + threadIdx.x) >> 5;
    const int lane = threadIdx.x & 31;
    const int nst  = N >> 5;                 // number of 32-wide strips
    const int mt   = wave / nst;
    const int ns   = wave - mt * nst;
    const int m0   = mt << 4;
    if (m0 >= M) return;
    const int n0   = ns << 5;
    const int row  = lane & 15;              // M in tile (A) / N in tile (B,C,D)
    const int hi   = lane >> 4;              // lane-group: selects K/M halves

    // clamp A row for ragged M edge (stores are guarded below)
    const int arow = (m0 + row < M) ? (m0 + row) : (M - 1);

    const bf16* Abase = A  + (size_t)arow * K;
    const bf16* Bc0   = Bt + (size_t)(n0 + row) * K;        // column for tile 0
    const bf16* Bc1   = Bt + (size_t)(n0 + 16 + row) * K;   // column for tile 1

    v8f acc0 = {}, acc1 = {};
    for (int k0 = 0; k0 < K; k0 += 32) {
        // A fragment (16x32): lanes 0-15 hold K=0..15, lanes 16-31 K=16..31.
        const v8bf* Ap = (const v8bf*)(Abase + k0 + hi * 16);
        v16bf a = __builtin_shufflevector(Ap[0], Ap[1],
                    0,1,2,3,4,5,6,7,8,9,10,11,12,13,14,15);
        // B fragment (32x16): lanes 0-15 hold K=0..15, lanes 16-31 K=16..31;
        // 16 contiguous bf16 per lane thanks to the transposed W.
        const v8bf* Bp0 = (const v8bf*)(Bc0 + k0 + hi * 16);
        v16bf b0 = __builtin_shufflevector(Bp0[0], Bp0[1],
                    0,1,2,3,4,5,6,7,8,9,10,11,12,13,14,15);
        const v8bf* Bp1 = (const v8bf*)(Bc1 + k0 + hi * 16);
        v16bf b1 = __builtin_shufflevector(Bp1[0], Bp1[1],
                    0,1,2,3,4,5,6,7,8,9,10,11,12,13,14,15);

        acc0 = __builtin_amdgcn_wmma_f32_16x16x32_bf16(
                   false, a, false, b0, (short)0, acc0, false, false);
        acc1 = __builtin_amdgcn_wmma_f32_16x16x32_bf16(
                   false, a, false, b1, (short)0, acc1, false, false);
    }

    // C/D layout: VGPR j -> M = j (lanes 0-15) or 8+j (lanes 16-31); N = lane&15.
    const int mbase = m0 + hi * 8;
    float* Cp = C + (size_t)mbase * N + n0 + row;
#pragma unroll
    for (int j = 0; j < 8; ++j) {
        if (mbase + j < M) {
            Cp[(size_t)j * N]      = acc0[j];
            Cp[(size_t)j * N + 16] = acc1[j];
        }
    }
}

// ---------------- per-(node,head) attention logits --------------------------
// asrc[n,h] = dot(xl[n,h,:], a_src[h,:]) ; adst likewise. H=8, C=64 fixed.
__global__ void attn_scores(const float* __restrict__ xl,
                            const float* __restrict__ a_src,
                            const float* __restrict__ a_dst,
                            float* __restrict__ asrc,
                            float* __restrict__ adst, int n_nodes) {
    int idx = blockIdx.x * blockDim.x + threadIdx.x;   // n*8 + h
    if (idx >= n_nodes * 8) return;
    int h = idx & 7;
    const float* xp = xl + (size_t)(idx >> 3) * 512 + h * 64;
    const float* as = a_src + h * 64;
    const float* ad = a_dst + h * 64;
    float s1 = 0.f, s2 = 0.f;
#pragma unroll 8
    for (int c = 0; c < 64; ++c) {
        float v = xp[c];
        s1 = fmaf(v, as[c], s1);
        s2 = fmaf(v, ad[c], s2);
    }
    asrc[idx] = s1;
    adst[idx] = s2;
}

// ---------------- CSR build: count -> scan -> fill --------------------------
__global__ void edge_count(const int* __restrict__ ei, int* __restrict__ deg,
                           int n_edges, int n_nodes) {
    int e = blockIdx.x * blockDim.x + threadIdx.x;
    if (e >= n_edges + n_nodes) return;
    int d = (e < n_edges) ? ei[n_edges + e] : (e - n_edges);   // self loops appended
    atomicAdd(&deg[d], 1);
}

__global__ void scan_exclusive(const int* __restrict__ deg,
                               int* __restrict__ rowptr,
                               int* __restrict__ cursor, int n) {
    __shared__ int tmp[256];
    __shared__ int carry;
    int tid = threadIdx.x;
    if (tid == 0) carry = 0;
    __syncthreads();
    for (int base = 0; base < n; base += 256) {
        int v = (base + tid < n) ? deg[base + tid] : 0;
        tmp[tid] = v;
        __syncthreads();
        for (int off = 1; off < 256; off <<= 1) {
            int t = (tid >= off) ? tmp[tid - off] : 0;
            __syncthreads();
            tmp[tid] += t;
            __syncthreads();
        }
        int excl = carry + tmp[tid] - v;
        if (base + tid < n) { rowptr[base + tid] = excl; cursor[base + tid] = excl; }
        __syncthreads();
        if (tid == 0) carry += tmp[255];
        __syncthreads();
    }
    if (tid == 0) rowptr[n] = carry;
}

__global__ void edge_fill(const int* __restrict__ ei, int* __restrict__ cursor,
                          int* __restrict__ csr_src, int n_edges, int n_nodes) {
    int e = blockIdx.x * blockDim.x + threadIdx.x;
    if (e >= n_edges + n_nodes) return;
    int s, d;
    if (e < n_edges) { s = ei[e]; d = ei[n_edges + e]; }
    else             { s = d = e - n_edges; }
    int pos = atomicAdd(&cursor[d], 1);
    csr_src[pos] = s;
}

// ---------------- segment-softmax statistics (per node,head) ----------------
__global__ void softmax_stats(const float* __restrict__ asrc,
                              const float* __restrict__ adst,
                              const int* __restrict__ rowptr,
                              const int* __restrict__ csr_src,
                              float* __restrict__ amax,
                              float* __restrict__ denom, int n_nodes) {
    int idx = blockIdx.x * blockDim.x + threadIdx.x;   // n*8 + h
    if (idx >= n_nodes * 8) return;
    int n = idx >> 3, h = idx & 7;
    float ad = adst[idx];
    int s0 = rowptr[n], s1 = rowptr[n + 1];
    float m = -__builtin_inff();
    for (int e = s0; e < s1; ++e) {
        float al = asrc[csr_src[e] * 8 + h] + ad;
        al = (al > 0.f) ? al : NEG_SLOPE * al;
        m = fmaxf(m, al);
    }
    if (!(m > -__builtin_inff())) m = 0.f;   // empty-segment guard (ref: where isfinite)
    float sum = 0.f;
    for (int e = s0; e < s1; ++e) {
        float al = asrc[csr_src[e] * 8 + h] + ad;
        al = (al > 0.f) ? al : NEG_SLOPE * al;
        sum += __expf(al - m);
    }
    amax[idx] = m;
    denom[idx] = sum;
}

// ---------------- fused aggregate + head-mean + bias (+relu) ----------------
// One block (256 threads, 8 waves) owns one destination node; no atomics.
// Thread t accumulates xl columns {t, t+256} (same channel, heads h and h+4),
// then the LDS reduction folds the 8 heads into the 64-channel output.
__global__ void aggregate(const float* __restrict__ xl,
                          const float* __restrict__ asrc,
                          const float* __restrict__ adst,
                          const float* __restrict__ amax,
                          const float* __restrict__ denom,
                          const int* __restrict__ rowptr,
                          const int* __restrict__ csr_src,
                          const float* __restrict__ bias,
                          float* __restrict__ out, int relu_flag) {
    __shared__ float s_coef[32 * 8];
    __shared__ int   s_src[32];
    __shared__ float s_red[256];

    const int n   = blockIdx.x;
    const int tid = threadIdx.x;
    const int start = rowptr[n], end = rowptr[n + 1];
    const int j1 = tid, j2 = tid + 256;
    const int h1 = j1 >> 6, h2 = j2 >> 6;
    const int el = tid >> 3, hh = tid & 7;      // coef producer mapping

    const float adn = adst[n * 8 + hh];
    const float mh  = amax[n * 8 + hh];
    const float dh  = denom[n * 8 + hh] + 1e-16f;

    float acc1 = 0.f, acc2 = 0.f;
    for (int base = start; base < end; base += 32) {
        const int cnt = min(32, end - base);
        if (el < cnt) {
            int s = csr_src[base + el];
            if (hh == 0) s_src[el] = s;
            float al = asrc[s * 8 + hh] + adn;
            al = (al > 0.f) ? al : NEG_SLOPE * al;
            s_coef[el * 8 + hh] = __expf(al - mh) / dh;
        }
        __syncthreads();
        for (int k = 0; k < cnt; ++k) {
            const float* xp = xl + (size_t)s_src[k] * 512;
            if (k + 1 < cnt)   // prefetch next gathered row (global_prefetch_b8)
                __builtin_prefetch(xl + (size_t)s_src[k + 1] * 512 + j1, 0, 0);
            acc1 = fmaf(xp[j1], s_coef[k * 8 + h1], acc1);
            acc2 = fmaf(xp[j2], s_coef[k * 8 + h2], acc2);
        }
        __syncthreads();
    }

    s_red[tid] = acc1 + acc2;   // heads {h1, h1+4} of channel tid&63
    __syncthreads();
    if (tid < 64) {
        float v = 0.125f * (s_red[tid] + s_red[tid + 64] +
                            s_red[tid + 128] + s_red[tid + 192]) + bias[tid];
        if (relu_flag) v = fmaxf(v, 0.f);
        out[(size_t)n * 64 + tid] = v;
    }
}

// ---------------------------------------------------------------------------
extern "C" void kernel_launch(void* const* d_in, const int* in_sizes, int n_in,
                              void* d_out, int out_size, void* d_ws, size_t ws_size,
                              hipStream_t stream) {
    const float* x      = (const float*)d_in[0];
    const int*   ei     = (const int*)  d_in[1];
    const float* W1     = (const float*)d_in[2];
    const float* a_src1 = (const float*)d_in[3];
    const float* a_dst1 = (const float*)d_in[4];
    const float* b1     = (const float*)d_in[5];
    const float* W2     = (const float*)d_in[6];
    const float* a_src2 = (const float*)d_in[7];
    const float* a_dst2 = (const float*)d_in[8];
    const float* b2     = (const float*)d_in[9];
    float* out = (float*)d_out;

    const int HC       = in_sizes[3];          // HEADS * h_feats = 512
    const int in_feats = in_sizes[2] / HC;     // 256
    const int n_nodes  = in_sizes[0] / in_feats;
    const int n_edges  = in_sizes[1] / 2;
    const int E_tot    = n_edges + n_nodes;    // with self loops

    // ---------------- workspace carve-up (float-sized slots, 16B aligned) ---
    float* wsf = (float*)d_ws;
    size_t o = 0;
    float* xl    = wsf + o; o += (size_t)n_nodes * 512;
    float* asrc  = wsf + o; o += (size_t)n_nodes * 8;
    float* adst  = wsf + o; o += (size_t)n_nodes * 8;
    float* amax  = wsf + o; o += (size_t)n_nodes * 8;
    float* denom = wsf + o; o += (size_t)n_nodes * 8;
    float* h1    = wsf + o; o += (size_t)n_nodes * 64;
    bf16* xbf    = (bf16*)(wsf + o); o += ((size_t)n_nodes * in_feats + 1) / 2;
    bf16* hbf    = (bf16*)(wsf + o); o += ((size_t)n_nodes * 64 + 1) / 2;
    bf16* wtbf   = (bf16*)(wsf + o); o += ((size_t)in_feats * HC + 1) / 2;  // Wt[N][K]
    int* rowptr  = (int*)(wsf + o);  o += (size_t)n_nodes + 4;
    int* cursor  = (int*)(wsf + o);  o += (size_t)n_nodes;
    int* deg     = (int*)(wsf + o);  o += (size_t)n_nodes;
    int* csr_src = (int*)(wsf + o);  o += (size_t)E_tot;

    const int T = 256;
    const int gNH   = (n_nodes * 8 + T - 1) / T;
    const int gEdge = (E_tot + T - 1) / T;

    // ---------------- CSR build (shared by both layers) ----------------------
    hipMemsetAsync(deg, 0, (size_t)n_nodes * sizeof(int), stream);
    edge_count<<<gEdge, T, 0, stream>>>(ei, deg, n_edges, n_nodes);
    scan_exclusive<<<1, T, 0, stream>>>(deg, rowptr, cursor, n_nodes);
    edge_fill<<<gEdge, T, 0, stream>>>(ei, cursor, csr_src, n_edges, n_nodes);

    // ---------------- layer 1 ------------------------------------------------
    {
        int nx = n_nodes * in_feats, nw = in_feats * HC;
        cvt_f32_to_bf16<<<(nx + T - 1) / T, T, 0, stream>>>(x, xbf, nx);
        cvt_transpose_w<<<(nw + T - 1) / T, T, 0, stream>>>(W1, wtbf, in_feats, HC);
        int mt = (n_nodes + 15) / 16, nst = HC / 32;
        int waves = mt * nst, blocks = (waves * 32 + T - 1) / T;
        wmma_gemm_bf16<<<blocks, T, 0, stream>>>(xbf, wtbf, xl, n_nodes, in_feats, HC);
    }
    attn_scores<<<gNH, T, 0, stream>>>(xl, a_src1, a_dst1, asrc, adst, n_nodes);
    softmax_stats<<<gNH, T, 0, stream>>>(asrc, adst, rowptr, csr_src, amax, denom, n_nodes);
    aggregate<<<n_nodes, T, 0, stream>>>(xl, asrc, adst, amax, denom,
                                         rowptr, csr_src, b1, h1, /*relu=*/1);

    // ---------------- layer 2 ------------------------------------------------
    {
        int nh = n_nodes * 64, nw = 64 * HC;
        cvt_f32_to_bf16<<<(nh + T - 1) / T, T, 0, stream>>>(h1, hbf, nh);
        cvt_transpose_w<<<(nw + T - 1) / T, T, 0, stream>>>(W2, wtbf, 64, HC);
        int mt = (n_nodes + 15) / 16, nst = HC / 32;
        int waves = mt * nst, blocks = (waves * 32 + T - 1) / T;
        wmma_gemm_bf16<<<blocks, T, 0, stream>>>(hbf, wtbf, xl, n_nodes, 64, HC);
    }
    attn_scores<<<gNH, T, 0, stream>>>(xl, a_src2, a_dst2, asrc, adst, n_nodes);
    softmax_stats<<<gNH, T, 0, stream>>>(asrc, adst, rowptr, csr_src, amax, denom, n_nodes);
    aggregate<<<n_nodes, T, 0, stream>>>(xl, asrc, adst, amax, denom,
                                         rowptr, csr_src, b2, out, /*relu=*/0);
}
